// ExtendedSelfAttention_31164282700017
// MI455X (gfx1250) — matrix operand order repaired and checked by reading, not compiled, not run on hardware
//
#include <hip/hip_runtime.h>
#include <hip/hip_bf16.h>
#include <stdint.h>

typedef __bf16 bf16_t;
typedef __bf16 v16bf __attribute__((ext_vector_type(16)));
typedef __bf16 v8bf  __attribute__((ext_vector_type(8)));
typedef __bf16 v4bf  __attribute__((ext_vector_type(4)));
typedef float  v8f   __attribute__((ext_vector_type(8)));
typedef float  v4f   __attribute__((ext_vector_type(4)));

// Problem sizes (fixed by the reference)
#define MB_   4
#define SS_   2048
#define DD_   4096
#define MTOT  (MB_ * SS_)   // 8192 rows

// LDS tile geometry: per stage, A(128 rows x 64 bf16) + B(128 rows x 64 bf16).
// Row stride padded to 144B (128B data + 16B) -> ds_load_b128 bank starts at
// 36*r mod 64: conflict-free (2-cycle minimum) fragment reads.
#define BK          64
#define LDSSTRIDE   144
#define B_OFF       (128 * LDSSTRIDE)        // 18432
#define STAGE_BYTES (2 * 128 * LDSSTRIDE)    // 36864
#define NSTAGE      3                        // 110592 B total LDS

// ---------------------------------------------------------------------------
// CDNA5 async global->LDS copy (ASYNCcnt-tracked) via inline asm.
// lds_off is the wave-relative LDS byte offset (low 32 bits of generic addr).
// ---------------------------------------------------------------------------
__device__ __forceinline__ void async_copy_b128(unsigned lds_off, const void* gaddr) {
  asm volatile("global_load_async_to_lds_b128 %0, %1, off"
               :: "v"(lds_off), "v"(gaddr) : "memory");
}
// Wait until at most 8 async ops outstanding: in-order completion means the
// oldest stage (8 copies) has fully landed while the next stage stays in flight.
__device__ __forceinline__ void wait_async_le8() {
  asm volatile("s_wait_asynccnt 8" ::: "memory");
}

// ---------------------------------------------------------------------------
// f32 -> bf16 conversion, 4-wide vectorized, grid-stride
// ---------------------------------------------------------------------------
__global__ __launch_bounds__(256) void cvt_f32_to_bf16_v4(
    const float* __restrict__ in, bf16_t* __restrict__ out, long n4)
{
  long i = (long)blockIdx.x * blockDim.x + threadIdx.x;
  long stride = (long)gridDim.x * blockDim.x;
  const v4f* in4 = (const v4f*)in;
  v4bf* out4 = (v4bf*)out;
  for (; i < n4; i += stride) {
    v4f x = in4[i];
    v4bf y;
    y.x = (bf16_t)x.x; y.y = (bf16_t)x.y;
    y.z = (bf16_t)x.z; y.w = (bf16_t)x.w;
    out4[i] = y;
  }
}

// ---------------------------------------------------------------------------
// C = A(MxK bf16) * B(NxK bf16)^T + bias(N), f32 accumulation.
// Workgroup 256 threads = 8 waves, block tile 128x128, K-step 64,
// 3-stage circular LDS buffer filled by global_load_async_to_lds_b128
// (2-iteration prefetch distance). Wave tile 32x64 = 2x4 WMMA accumulators,
// 16 v_wmma_f32_16x16x32_bf16 per iteration per wave.
// ---------------------------------------------------------------------------
template <bool OUT_BF16>
__global__ __launch_bounds__(256) void gemm_bf16_nt_lds(
    const bf16_t* __restrict__ A,    // M x K
    const bf16_t* __restrict__ Bm,   // N x K
    const float*  __restrict__ bias, // N (may be null)
    void* __restrict__ Out,          // M x N (bf16 or f32)
    int M, int N, int K)
{
  __shared__ __align__(16) char smem[NSTAGE * STAGE_BYTES];

  const int tid  = threadIdx.x;
  const int lane = tid & 31;
  const int wave = tid >> 5;
  const int sel  = lane >> 4;   // K-half select within fragment (ISA layout)
  const int r    = lane & 15;   // row (A) / column (B) within 16

  const int wm = wave & 3;      // 0..3 -> 32-row strip
  const int wn = wave >> 2;     // 0..1 -> 64-col strip

  const int blockM = blockIdx.y * 128;
  const int blockN = blockIdx.x * 128;

  // ---- async copy addressing: 1024 16B chunks per matrix per stage,
  //      4 per thread per matrix (rows trow+32q, 16B column tcol) ----
  const int trow = tid >> 3;    // 0..31
  const int tcol = tid & 7;     // 0..7
  const unsigned sbase = (unsigned)(uintptr_t)&smem[0]; // wave-relative LDS offset

  const bf16_t* gA[4];
  const bf16_t* gB[4];
  unsigned lA[4], lB[4];
#pragma unroll
  for (int q = 0; q < 4; ++q) {
    gA[q] = A  + (size_t)(blockM + trow + 32 * q) * K + tcol * 8;
    gB[q] = Bm + (size_t)(blockN + trow + 32 * q) * K + tcol * 8;
    lA[q] = sbase + (trow + 32 * q) * LDSSTRIDE + tcol * 16;
    lB[q] = sbase + B_OFF + (trow + 32 * q) * LDSSTRIDE + tcol * 16;
  }

  auto copy_stage = [&](int s, int k0) {
    const unsigned so = (unsigned)(s * STAGE_BYTES);
#pragma unroll
    for (int q = 0; q < 4; ++q) {
      async_copy_b128(lA[q] + so, gA[q] + k0);
      async_copy_b128(lB[q] + so, gB[q] + k0);
    }
  };

  // ---- fragment read offsets into LDS ----
  // A frag i (slice ks): row = wm*32 + i*16 + r, bytes ks*64 + sel*16 (+0,+32)
  // B frag j (slice ks): row = wn*64 + j*16 + r, bytes ks*64 + sel*16 (+0,+32)
  const char* saf = smem + (wm * 32 + r) * LDSSTRIDE + sel * 16;
  const char* sbf = smem + B_OFF + (wn * 64 + r) * LDSSTRIDE + sel * 16;

  v8f acc[2][4];
#pragma unroll
  for (int i = 0; i < 2; ++i)
#pragma unroll
    for (int j = 0; j < 4; ++j)
      acc[i][j] = v8f{0.f, 0.f, 0.f, 0.f, 0.f, 0.f, 0.f, 0.f};

  // Prologue: two stages in flight (16 outstanding asyncs per wave).
  copy_stage(0, 0);
  copy_stage(1, (BK < K) ? BK : 0);

  int cur = 0;
  for (int k0 = 0; k0 < K; k0 += BK) {
    wait_async_le8();     // oldest stage (cur) fully landed; next stays in flight
    __syncthreads();      // all waves landed stage cur; prior reads of the
                          // buffer we are about to refill are complete

    // Always issue (uniform ASYNCcnt bookkeeping); redirect the tail to k=0
    // into a buffer that is never read again.
    int knext = k0 + 2 * BK;
    if (knext >= K) knext = 0;
    int snext = cur + 2; if (snext >= NSTAGE) snext -= NSTAGE;
    copy_stage(snext, knext);

    const char* sa = saf + cur * STAGE_BYTES;
    const char* sb = sbf + cur * STAGE_BYTES;

    union Frag { v16bf v; v8bf h[2]; };
#pragma unroll
    for (int ks = 0; ks < 2; ++ks) {          // two 32-wide K slices per stage
      Frag fa[2], fb[4];
#pragma unroll
      for (int i = 0; i < 2; ++i) {
        fa[i].h[0] = *(const v8bf*)(sa + ks * 64 + i * 16 * LDSSTRIDE);
        fa[i].h[1] = *(const v8bf*)(sa + ks * 64 + i * 16 * LDSSTRIDE + 32);
      }
#pragma unroll
      for (int j = 0; j < 4; ++j) {
        fb[j].h[0] = *(const v8bf*)(sb + ks * 64 + j * 16 * LDSSTRIDE);
        fb[j].h[1] = *(const v8bf*)(sb + ks * 64 + j * 16 * LDSSTRIDE + 32);
      }
#pragma unroll
      for (int i = 0; i < 2; ++i)
#pragma unroll
        for (int j = 0; j < 4; ++j)
          acc[i][j] = __builtin_amdgcn_wmma_f32_16x16x32_bf16(
              false, fa[i].v, false, fb[j].v, (short)0, acc[i][j], false, false);
    }

    cur = cur + 1; if (cur >= NSTAGE) cur -= NSTAGE;
  }

  // ---- writeback: M = m0 + i*16 + sel*8 + t, N = n0 + j*16 + r ----
  const int m0 = blockM + wm * 32;
  const int n0 = blockN + wn * 64;
#pragma unroll
  for (int i = 0; i < 2; ++i) {
#pragma unroll
    for (int j = 0; j < 4; ++j) {
      const int n = n0 + j * 16 + r;
      const float badd = bias ? bias[n] : 0.f;
#pragma unroll
      for (int t = 0; t < 8; ++t) {
        const int m = m0 + i * 16 + sel * 8 + t;
        const float val = acc[i][j][t] + badd;
        if (OUT_BF16)
          ((bf16_t*)Out)[(long)m * N + n] = (bf16_t)val;
        else
          ((float*)Out)[(long)m * N + n] = val;
      }
    }
  }
}

// ---------------------------------------------------------------------------
// Launcher.  With H=1, softmax over the size-1 head axis == 1.0 exactly, so
//   out = (value @ Wv.T + bv) @ Wo.T + bo;  the q/k paths are dead code.
// Workspace layout (bytes):
//   [0,    64MB)  value_bf16  (8192*4096*2)
//   [64MB, 96MB)  Wv_bf16     (4096*4096*2)
//   [96MB,128MB)  Wo_bf16     (4096*4096*2)
//   [128MB,192MB) v_bf16      (8192*4096*2)
// ---------------------------------------------------------------------------
extern "C" void kernel_launch(void* const* d_in, const int* in_sizes, int n_in,
                              void* d_out, int out_size, void* d_ws, size_t ws_size,
                              hipStream_t stream)
{
  (void)in_sizes; (void)n_in; (void)out_size; (void)ws_size;

  const float* value = (const float*)d_in[2];
  const float* bv    = (const float*)d_in[8];
  const float* Wv    = (const float*)d_in[7];
  const float* Wo    = (const float*)d_in[9];
  const float* bo    = (const float*)d_in[10];

  const int M = MTOT;   // 8192
  const int D = DD_;    // 4096

  char* ws = (char*)d_ws;
  bf16_t* value_bf = (bf16_t*)(ws);
  bf16_t* Wv_bf    = (bf16_t*)(ws + (size_t)M * D * 2);
  bf16_t* Wo_bf    = (bf16_t*)(ws + (size_t)M * D * 2 + (size_t)D * D * 2);
  bf16_t* v_bf     = (bf16_t*)(ws + (size_t)M * D * 2 + 2 * (size_t)D * D * 2);

  cvt_f32_to_bf16_v4<<<2048, 256, 0, stream>>>(value, value_bf, (long)M * D / 4);
  cvt_f32_to_bf16_v4<<<1024, 256, 0, stream>>>(Wv, Wv_bf, (long)D * D / 4);
  cvt_f32_to_bf16_v4<<<1024, 256, 0, stream>>>(Wo, Wo_bf, (long)D * D / 4);

  dim3 grid(D / 128, M / 128);  // (32, 64)
  dim3 block(256);

  // GEMM1: v = value @ Wv.T + bv  (bf16 out, f32 accum)
  gemm_bf16_nt_lds<true ><<<grid, block, 0, stream>>>(value_bf, Wv_bf, bv, (void*)v_bf, M, D, D);
  // GEMM2: out = v @ Wo.T + bo    (f32 out)
  gemm_bf16_nt_lds<false><<<grid, block, 0, stream>>>(v_bf, Wo_bf, bo, d_out, M, D, D);
}
